// LSTMModel_77515569758624
// MI455X (gfx1250) — compile-verified, matrix operand order checked by reading, not run on hardware
//
#include <hip/hip_runtime.h>

typedef __attribute__((ext_vector_type(16))) __bf16 v16bf;
typedef __attribute__((ext_vector_type(8)))  float  v8f;

union V16U { v16bf v; unsigned u[8]; };
union V8FU { v8f v; float f[8]; };

constexpr int kB = 1024, kT = 512, kI = 6, kH = 64, kG = 256, kMT = 16, kO = 2;

__device__ __forceinline__ unsigned pk_bf16(float a, float b) {
  unsigned short ua = __builtin_bit_cast(unsigned short, (__bf16)a);
  unsigned short ub = __builtin_bit_cast(unsigned short, (__bf16)b);
  return (unsigned)ua | ((unsigned)ub << 16);
}

// Unified activation: sigmoid(x) = rcp(1+exp(-x)); tanh(x) = 2*sigmoid(2x)-1.
// (a, m, b) are wave-uniform; v_exp + v_rcp + fma, fully branchless.
__device__ __forceinline__ float act_unified(float x, float a, float m, float b) {
  float e = __expf(-a * x);                      // v_mul + v_exp_f32
  float s = __builtin_amdgcn_rcpf(1.0f + e);     // v_add + v_rcp_f32
  return __builtin_fmaf(s, m, b);                // v_fma
}
__device__ __forceinline__ float fast_tanh(float v) {
  // 1 - 2*rcp(exp(2v)+1): robust at both extremes, no IEEE divide.
  float e = __expf(2.0f * v);
  return __builtin_fmaf(-2.0f, __builtin_amdgcn_rcpf(e + 1.0f), 1.0f);
}

__global__ __launch_bounds__(256) void lstm_fused(
    const float* __restrict__ x, const float* __restrict__ W_ih,
    const float* __restrict__ W_hh, const float* __restrict__ b_ih,
    const float* __restrict__ b_hh, const float* __restrict__ W_fc,
    const float* __restrict__ b_fc, float* __restrict__ out) {
  // LDS: ~23 KB total
  __shared__ __align__(16) __bf16 hbf[kMT * kH];   // h_t bf16, row-major [m][k]
  __shared__ __align__(16) __bf16 xbf[kMT * 8];    // x_t bf16, [m][8], slots 6,7 = 0
  __shared__ __align__(16) float  gbuf[kG * kMT];  // activated gates, [n][m]
  __shared__ __align__(16) float  hf32[kMT * kH];  // final h (fp32) for FC
  __shared__ float wfc[kO * kH + kO];

  const int tid   = threadIdx.x;
  const int wv    = tid >> 5;      // wave id 0..7
  const int lane  = tid & 31;
  const int lh    = lane >> 4;     // lane half (WMMA operand layout)
  const int ln    = lane & 15;
  const int bbase = blockIdx.x * kMT;

  // ---- one-time init ----
  for (int i = tid; i < kMT * kH / 2; i += 256) ((unsigned*)hbf)[i] = 0u;  // h0 = 0
  if (tid < kMT * 2) {                       // zero x pad slots (K=6,7), never touched again
    int r = tid >> 1, j = 6 + (tid & 1);
    xbf[r * 8 + j] = (__bf16)0.0f;
  }
  if (tid < kO * kH) wfc[tid] = W_fc[tid];
  if (tid < kO)      wfc[kO * kH + tid] = b_fc[tid];
  if (tid < kMT * kI) {                      // stage x for t=0
    int r = tid / kI, i = tid % kI;
    xbf[r * 8 + i] = (__bf16)x[(size_t)(bbase + r) * kT * kI + i];
  }

  // ---- per-wave persistent operands (registers for the whole scan) ----
  const int n0 = (2 * wv) * 16 + ln;         // gate columns handled by this lane
  const int n1 = (2 * wv + 1) * 16 + ln;
  const float bias0 = b_ih[n0] + b_hh[n0];
  const float bias1 = b_ih[n1] + b_hh[n1];

  V16U Bh[2][2];  // [n-tile][k-chunk]: W_hh bf16 in B-operand layout
  V16U Bx[2];     // W_ih bf16 (K<6 live, rest zero)
  {
    const float* rows[2] = { W_hh + (size_t)n0 * kH, W_hh + (size_t)n1 * kH };
    for (int tl = 0; tl < 2; ++tl) {
      for (int c = 0; c < 2; ++c) {
        const int ka = 32 * c + lh * 8;       // VGPRs 0-3: K = c*32 + lh*8 + 0..7
        const int kb = 32 * c + 16 + lh * 8;  // VGPRs 4-7: K = c*32 + 16 + lh*8 + 0..7
        for (int j = 0; j < 4; ++j) {
          Bh[tl][c].u[j]     = pk_bf16(rows[tl][ka + 2 * j], rows[tl][ka + 2 * j + 1]);
          Bh[tl][c].u[4 + j] = pk_bf16(rows[tl][kb + 2 * j], rows[tl][kb + 2 * j + 1]);
        }
      }
    }
    const int nn[2] = { n0, n1 };
    for (int tl = 0; tl < 2; ++tl) {
      for (int j = 0; j < 8; ++j) Bx[tl].u[j] = 0u;
      if (lh == 0) {                           // K=0..5 only
        const float* wr = W_ih + (size_t)nn[tl] * kI;
        Bx[tl].u[0] = pk_bf16(wr[0], wr[1]);
        Bx[tl].u[1] = pk_bf16(wr[2], wr[3]);
        Bx[tl].u[2] = pk_bf16(wr[4], wr[5]);
      }
    }
  }

  // Wave-uniform activation constants: waves 4,5 hold the "g" gate -> tanh.
  const bool  gate_tanh = ((wv >> 1) == 2);
  const float act_a = gate_tanh ? 2.0f : 1.0f;   // input scale
  const float act_m = gate_tanh ? 2.0f : 1.0f;   // output scale
  const float act_b = gate_tanh ? -1.0f : 0.0f;  // output offset
  const unsigned xmask = (lh == 0) ? 0xFFFFFFFFu : 0u;  // branchless Ax masking

  const int  ek = tid & 63;                  // elementwise phase: hidden index
  const int  em = (tid >> 6) << 2;           // elementwise phase: 4 batch rows
  float cst[4] = {0.f, 0.f, 0.f, 0.f};       // cell state lives in registers

  __syncthreads();

  for (int t = 0; t < kT; ++t) {
    // ---------- Phase A: gates = bias + x_t W_ih^T + h_t W_hh^T (WMMA) ----------
    V8FU acc0, acc1;
    for (int j = 0; j < 8; ++j) { acc0.f[j] = bias0; acc1.f[j] = bias1; }

    V16U A0, A1, Ax;
    {
      const __bf16* hr = hbf + ln * kH;      // A layout: lane -> row m = ln
      uint4 q0 = *(const uint4*)(hr + lh * 8);
      uint4 q1 = *(const uint4*)(hr + 16 + lh * 8);
      uint4 q2 = *(const uint4*)(hr + 32 + lh * 8);
      uint4 q3 = *(const uint4*)(hr + 48 + lh * 8);
      A0.u[0] = q0.x; A0.u[1] = q0.y; A0.u[2] = q0.z; A0.u[3] = q0.w;
      A0.u[4] = q1.x; A0.u[5] = q1.y; A0.u[6] = q1.z; A0.u[7] = q1.w;
      A1.u[0] = q2.x; A1.u[1] = q2.y; A1.u[2] = q2.z; A1.u[3] = q2.w;
      A1.u[4] = q3.x; A1.u[5] = q3.y; A1.u[6] = q3.z; A1.u[7] = q3.w;
      uint4 xq = *(const uint4*)(xbf + ln * 8);        // K=0..7 (6,7 are zero)
      Ax.u[0] = xq.x & xmask; Ax.u[1] = xq.y & xmask;  // upper lane-half: K=8..15 -> 0
      Ax.u[2] = xq.z & xmask; Ax.u[3] = xq.w & xmask;
      Ax.u[4] = 0u; Ax.u[5] = 0u; Ax.u[6] = 0u; Ax.u[7] = 0u;
    }

    acc0.v = __builtin_amdgcn_wmma_f32_16x16x32_bf16(false, Ax.v, false, Bx[0].v,    (short)0, acc0.v, false, false);
    acc0.v = __builtin_amdgcn_wmma_f32_16x16x32_bf16(false, A0.v, false, Bh[0][0].v, (short)0, acc0.v, false, false);
    acc0.v = __builtin_amdgcn_wmma_f32_16x16x32_bf16(false, A1.v, false, Bh[0][1].v, (short)0, acc0.v, false, false);
    acc1.v = __builtin_amdgcn_wmma_f32_16x16x32_bf16(false, Ax.v, false, Bx[1].v,    (short)0, acc1.v, false, false);
    acc1.v = __builtin_amdgcn_wmma_f32_16x16x32_bf16(false, A0.v, false, Bh[1][0].v, (short)0, acc1.v, false, false);
    acc1.v = __builtin_amdgcn_wmma_f32_16x16x32_bf16(false, A1.v, false, Bh[1][1].v, (short)0, acc1.v, false, false);

    // Branchless, divide-free activation (wave-uniform gate type).
    for (int j = 0; j < 8; ++j) {
      acc0.f[j] = act_unified(acc0.f[j], act_a, act_m, act_b);
      acc1.f[j] = act_unified(acc1.f[j], act_a, act_m, act_b);
    }
    // C/D layout: VGPR j -> row m = j + 8*lh, column n per-lane -> gbuf[n][m]
    *(float4*)&gbuf[n0 * kMT + lh * 8]     = *(float4*)&acc0.f[0];
    *(float4*)&gbuf[n0 * kMT + lh * 8 + 4] = *(float4*)&acc0.f[4];
    *(float4*)&gbuf[n1 * kMT + lh * 8]     = *(float4*)&acc1.f[0];
    *(float4*)&gbuf[n1 * kMT + lh * 8 + 4] = *(float4*)&acc1.f[4];

    __syncthreads();

    // ---------- Phase B: c' = f*c + i*g ; h' = o*tanh(c') ----------
    {
      const float* ip = &gbuf[(0 * kH + ek) * kMT + em];
      const float* fp = &gbuf[(1 * kH + ek) * kMT + em];
      const float* gp = &gbuf[(2 * kH + ek) * kMT + em];
      const float* op = &gbuf[(3 * kH + ek) * kMT + em];
#pragma unroll
      for (int jj = 0; jj < 4; ++jj) {
        float cn = __builtin_fmaf(fp[jj], cst[jj], ip[jj] * gp[jj]);
        float hn = op[jj] * fast_tanh(cn);
        cst[jj] = cn;
        hbf[(em + jj) * kH + ek] = (__bf16)hn;
        if (t == kT - 1) {                        // uniform branch, taken once
          hf32[(em + jj) * kH + ek] = hn;
          out[kB * kO + (size_t)(bbase + em + jj) * kH + ek] = hn;             // h_T
          out[kB * kO + kB * kH + (size_t)(bbase + em + jj) * kH + ek] = cn;   // c_T
        }
      }
      if (t + 1 < kT && tid < kMT * kI) {      // stage x for next step
        int r = tid / kI, i = tid % kI;
        xbf[r * 8 + i] = (__bf16)x[(size_t)(bbase + r) * kT * kI + (size_t)(t + 1) * kI + i];
      }
    }
    __syncthreads();
  }

  // ---------- FC on h_T: out[b,o] = h . W_fc[o] + b_fc[o] ----------
  if (tid < kMT * kO) {
    const int m = tid >> 1, o = tid & 1;
    float s = wfc[kO * kH + o];
    const float* wr = &wfc[o * kH];
    const float* hv = &hf32[m * kH];
    for (int k2 = 0; k2 < kH; ++k2) s = __builtin_fmaf(hv[k2], wr[k2], s);
    out[(size_t)(bbase + m) * kO + o] = s;
  }
}

extern "C" void kernel_launch(void* const* d_in, const int* in_sizes, int n_in,
                              void* d_out, int out_size, void* d_ws, size_t ws_size,
                              hipStream_t stream) {
  (void)in_sizes; (void)n_in; (void)d_ws; (void)ws_size; (void)out_size;
  const float* x    = (const float*)d_in[0];
  const float* W_ih = (const float*)d_in[1];
  const float* W_hh = (const float*)d_in[2];
  const float* b_ih = (const float*)d_in[3];
  const float* b_hh = (const float*)d_in[4];
  const float* W_fc = (const float*)d_in[5];
  const float* b_fc = (const float*)d_in[6];
  float* out = (float*)d_out;

  dim3 grid(kB / kMT);   // 64 batch tiles of 16 rows, independent recurrences
  dim3 block(256);       // 8 wave32 waves per workgroup
  hipLaunchKernelGGL(lstm_fused, grid, block, 0, stream,
                     x, W_ih, W_hh, b_ih, b_hh, W_fc, b_fc, out);
}
